// GraphModel_46334107189742
// MI455X (gfx1250) — matrix operand order repaired and checked by reading, not compiled, not run on hardware
//
#include <hip/hip_runtime.h>

typedef __attribute__((ext_vector_type(16))) _Float16 v16h;
typedef __attribute__((ext_vector_type(8)))  _Float16 v8h;
typedef __attribute__((ext_vector_type(8)))  float    v8f;

#define HC 640
#define NH 5

__device__ __forceinline__ float leaky02(float x){ return x > 0.f ? x : 0.2f*x; }
// order-preserving float<->uint encoding for atomicMax on floats
__device__ __forceinline__ unsigned encf(float f){ unsigned u=__float_as_uint(f); return (u&0x80000000u)?~u:(u|0x80000000u); }
__device__ __forceinline__ float decf(unsigned e){ return (e&0x80000000u)?__uint_as_float(e^0x80000000u):__uint_as_float(~e); }

// ---------------- f32 -> f16 conversion with K padding (A: [rows,K] -> [rows,Kp]) ----
__global__ void cvt_pad_a(const float* __restrict__ src, _Float16* __restrict__ dst,
                          int rows, int K, int Kp) {
  int i = blockIdx.x * blockDim.x + threadIdx.x;
  if (i >= rows * Kp) return;
  int r = i / Kp, k = i - r * Kp;
  dst[i] = (k < K) ? (_Float16)src[r * K + k] : (_Float16)0.f;
}
// W: [K,Nc] -> [Kp,Nc] f16, zero rows K..Kp
__global__ void cvt_pad_w(const float* __restrict__ W, _Float16* __restrict__ Wh,
                          int K, int Kp, int Nc) {
  int i = blockIdx.x * blockDim.x + threadIdx.x;
  if (i >= Kp * Nc) return;
  int k = i / Nc;
  Wh[i] = (k < K) ? (_Float16)W[i] : (_Float16)0.f;
}

// ---------------- WMMA GEMM: C[M,Nc] = A[M,Kp] @ B[Kp,Nc], f16 in / f32 out ----
// one wave per 16(M) x 64(N) strip; Kp % 32 == 0 (pre-padded, no guards).
// A frag: ISA 7.12.2 16-bit A 16x32 -> two contiguous 8-half runs per lane.
// B frag: lane%16 = N column, element j = K row in A's k order -> strided 2B loads.
__global__ void gemm_wmma(const _Float16* __restrict__ A, const _Float16* __restrict__ B,
                          float* __restrict__ C, int M, int Kp, int Nc) {
  int wave = blockIdx.x * (blockDim.x >> 5) + (threadIdx.x >> 5);
  int lane = threadIdx.x & 31;
  int tilesN = Nc >> 6;                       // 64-wide strips
  int tileM = wave / tilesN, tileN = wave - tileM * tilesN;
  if (tileM * 16 >= M) return;
  int m  = tileM * 16 + (lane & 15);
  int n0 = tileN * 64;
  bool hi = lane >= 16;
  v8f acc0 = {}, acc1 = {}, acc2 = {}, acc3 = {};
  const _Float16* Arow = A + (size_t)m * Kp + (hi ? 8 : 0);
  for (int k0 = 0; k0 < Kp; k0 += 32) {
    v8h alo = *(const v8h*)(Arow + k0);        // k = k0+base+0..7
    v8h ahi = *(const v8h*)(Arow + k0 + 16);   // k = k0+base+16..23
    v16h a;
#pragma unroll
    for (int j = 0; j < 8; ++j) { a[j] = alo[j]; a[j + 8] = ahi[j]; }
    const _Float16* Brow = B + (size_t)k0 * Nc + n0 + (lane & 15);
    v16h b0, b1, b2, b3;
#pragma unroll
    for (int j = 0; j < 16; ++j) {
      int kk = (j < 8) ? ((hi ? 8 : 0) + j) : (16 + (hi ? 8 : 0) + (j - 8));
      const _Float16* q = Brow + (size_t)kk * Nc;
      b0[j] = q[0]; b1[j] = q[16]; b2[j] = q[32]; b3[j] = q[48];
    }
    acc0 = __builtin_amdgcn_wmma_f32_16x16x32_f16(false, a, false, b0, (short)0, acc0, false, false);
    acc1 = __builtin_amdgcn_wmma_f32_16x16x32_f16(false, a, false, b1, (short)0, acc1, false, false);
    acc2 = __builtin_amdgcn_wmma_f32_16x16x32_f16(false, a, false, b2, (short)0, acc2, false, false);
    acc3 = __builtin_amdgcn_wmma_f32_16x16x32_f16(false, a, false, b3, (short)0, acc3, false, false);
  }
  int row0 = tileM * 16 + (hi ? 8 : 0);
  int col  = lane & 15;
#pragma unroll
  for (int r = 0; r < 8; ++r) {
    float* Crow = C + (size_t)(row0 + r) * Nc + col;
    Crow[n0]      = acc0[r];
    Crow[n0 + 16] = acc1[r];
    Crow[n0 + 32] = acc2[r];
    Crow[n0 + 48] = acc3[r];
  }
}

// ---------------- CSR build (once): degree, scan, fill ----------------
__global__ void count_deg(const int* __restrict__ dst, int* deg, int E) {
  int e = blockIdx.x * blockDim.x + threadIdx.x;
  if (e < E) atomicAdd(&deg[dst[e]], 1);
}
__global__ void scan_deg(const int* __restrict__ deg, int* __restrict__ rowstart, int N) {
  __shared__ int s[1024];
  __shared__ int running;
  int tid = threadIdx.x;
  if (tid == 0) running = 0;
  __syncthreads();
  for (int base = 0; base < N; base += 1024) {
    int i = base + tid;
    int v = (i < N) ? deg[i] : 0;
    s[tid] = v; __syncthreads();
    for (int off = 1; off < 1024; off <<= 1) {
      int t = (tid >= off) ? s[tid - off] : 0;
      __syncthreads();
      s[tid] += t;
      __syncthreads();
    }
    int br = running;
    if (i < N) rowstart[i] = br + s[tid] - v;
    __syncthreads();
    if (tid == 0) running = br + s[1023];
    __syncthreads();
  }
  if (tid == 0) rowstart[N] = running;
}
__global__ void csr_fill(const int* __restrict__ dst, const int* __restrict__ rowstart,
                         int* fillpos, int* csr, int E) {
  int e = blockIdx.x * blockDim.x + threadIdx.x;
  if (e < E) { int d = dst[e]; csr[rowstart[d] + atomicAdd(&fillpos[d], 1)] = e; }
}

// ---------------- fold We with a_edge: wfold[10x5] ----------------
__global__ void fold_we(const float* __restrict__ We, const float* __restrict__ aedge,
                        float* wfold) {
  int i = threadIdx.x;
  if (i >= 50) return;
  int k = i / NH, h = i - k * NH;
  float s = 0.f;
  for (int c = 0; c < 128; ++c) s += We[k * HC + h * 128 + c] * aedge[h * 128 + c];
  wfold[i] = s;
}

// ---------------- a_l, a_r per node-head (wave per node) ----------------
__global__ void node_alr(const float* __restrict__ h, const float* __restrict__ asrc,
                         const float* __restrict__ adst, float* al, float* ar, int N) {
  int n = blockIdx.x * (blockDim.x >> 5) + (threadIdx.x >> 5);
  int lane = threadIdx.x & 31;
  if (n >= N) return;
  for (int hh = 0; hh < NH; ++hh) {
    float sl = 0.f, sr = 0.f;
    for (int i = 0; i < 4; ++i) {
      int c = hh * 128 + i * 32 + lane;
      float v = h[n * HC + c];
      sl += v * asrc[c];
      sr += v * adst[c];
    }
    for (int off = 16; off; off >>= 1) { sl += __shfl_xor(sl, off, 32); sr += __shfl_xor(sr, off, 32); }
    if (lane == 0) { al[n * NH + hh] = sl; ar[n * NH + hh] = sr; }
  }
}

// ---------------- per-edge folded attention score + loop-attr accumulation ----
__global__ void edge_ae(const float* __restrict__ ea, const float* __restrict__ wfold,
                        const int* __restrict__ dst, float* aebuf, float* loopsum, int E) {
  int e = blockIdx.x * blockDim.x + threadIdx.x;
  if (e >= E) return;
  float f[10];
#pragma unroll
  for (int k = 0; k < 10; ++k) f[k] = ea[e * 10 + k];
  int d = dst[e];
#pragma unroll
  for (int hh = 0; hh < NH; ++hh) {
    float s = 0.f;
#pragma unroll
    for (int k = 0; k < 10; ++k) s += f[k] * wfold[k * NH + hh];
    aebuf[e * NH + hh] = s;
    atomicAdd(&loopsum[d * NH + hh], s);
  }
}

// ---------------- self-loop alpha, init amax/denom ----------------
__global__ void node_loop(const float* __restrict__ al, const float* __restrict__ ar,
                          const float* __restrict__ loopsum, const int* __restrict__ deg,
                          float* loop_alpha, unsigned* amax, float* denom, int N5) {
  int i = blockIdx.x * blockDim.x + threadIdx.x;
  if (i >= N5) return;
  int n = i / NH;
  float dg = fmaxf((float)deg[n], 1.f);
  float la = leaky02(al[i] + ar[i] + loopsum[i] / dg);
  loop_alpha[i] = la;
  amax[i] = encf(la);
  denom[i] = 0.f;
}

// ---------------- per-edge alpha + segment max ----------------
__global__ void edge_alpha(const float* __restrict__ al, const float* __restrict__ ar,
                           const int* __restrict__ src, const int* __restrict__ dst,
                           float* aebuf, unsigned* amax, int E5) {
  int i = blockIdx.x * blockDim.x + threadIdx.x;
  if (i >= E5) return;
  int e = i / NH, hh = i - e * NH;
  int s = src[e], d = dst[e];
  float a = leaky02(al[s * NH + hh] + ar[d * NH + hh] + aebuf[i]);
  aebuf[i] = a;
  atomicMax(&amax[d * NH + hh], encf(a));
}

// ---------------- per-edge exp + segment sum ----------------
__global__ void edge_exp(float* aebuf, const int* __restrict__ dst,
                         const unsigned* __restrict__ amax, float* denom, int E5) {
  int i = blockIdx.x * blockDim.x + threadIdx.x;
  if (i >= E5) return;
  int e = i / NH, hh = i - e * NH;
  int d = dst[e];
  float ex = expf(aebuf[i] - decf(amax[d * NH + hh]));
  aebuf[i] = ex;
  atomicAdd(&denom[d * NH + hh], ex);
}

// ---------------- finalize softmax: inv-denom and loop attention ----------------
__global__ void node_norm(float* loop_alpha, const unsigned* __restrict__ amax,
                          float* denom, int N5) {
  int i = blockIdx.x * blockDim.x + threadIdx.x;
  if (i >= N5) return;
  float exl = expf(loop_alpha[i] - decf(amax[i]));
  float inv = 1.f / (denom[i] + exl + 1e-16f);
  denom[i]      = inv;        // now inv_denom
  loop_alpha[i] = exl * inv;  // now att_loop
}

// ---------------- CSR gather-aggregation: block(160) per node, wave per head ----
__global__ void aggregate(const float* __restrict__ h, const int* __restrict__ src,
                          const int* __restrict__ rowstart, const int* __restrict__ csr,
                          const float* __restrict__ ex, const float* __restrict__ inv,
                          const float* __restrict__ attloop, const float* __restrict__ bias,
                          float* __restrict__ out, int relu, int N) {
  int n = blockIdx.x;
  int t = threadIdx.x;          // 0..159
  int c = t * 4;                // channel base
  int hh = t >> 5;              // head = wave index
  float ax = 0.f, ay = 0.f, az = 0.f, aw = 0.f;
  float winv = inv[n * NH + hh];
  int beg = rowstart[n], end = rowstart[n + 1];
  for (int p = beg; p < end; ++p) {
    int e = csr[p];
    int s = src[e];
    float w = ex[e * NH + hh] * winv;
    const float4 hv = *(const float4*)(h + s * HC + c);
    ax += hv.x * w; ay += hv.y * w; az += hv.z * w; aw += hv.w * w;
  }
  float wl = attloop[n * NH + hh];
  const float4 hs = *(const float4*)(h + n * HC + c);
  ax += hs.x * wl; ay += hs.y * wl; az += hs.z * wl; aw += hs.w * wl;
  const float4 bv = *(const float4*)(bias + c);
  ax += bv.x; ay += bv.y; az += bv.z; aw += bv.w;
  if (relu) { ax = fmaxf(ax, 0.f); ay = fmaxf(ay, 0.f); az = fmaxf(az, 0.f); aw = fmaxf(aw, 0.f); }
  float4 o; o.x = ax; o.y = ay; o.z = az; o.w = aw;
  *(float4*)(out + n * HC + c) = o;
}

// ---------------- global mean pool + linear head ----------------
__global__ void pool_cnt(const int* __restrict__ batch, int* cnt, int N) {
  int n = blockIdx.x * blockDim.x + threadIdx.x;
  if (n < N) atomicAdd(&cnt[batch[n]], 1);
}
__global__ void pool_sum(const float* __restrict__ hf, const int* __restrict__ batch,
                         float* pooled, int NC) {
  int i = blockIdx.x * blockDim.x + threadIdx.x;
  if (i >= NC) return;
  int n = i / HC, c = i - n * HC;
  atomicAdd(&pooled[batch[n] * HC + c], hf[i]);
}
__global__ void pool_final(const float* __restrict__ pooled, const int* __restrict__ cnt,
                           const float* __restrict__ Wlin, const float* __restrict__ blin,
                           float* out, int G) {
  int g = blockIdx.x * (blockDim.x >> 5) + (threadIdx.x >> 5);
  int lane = threadIdx.x & 31;
  if (g >= G) return;
  float s = 0.f;
  for (int c = lane; c < HC; c += 32) s += pooled[g * HC + c] * Wlin[c];
  for (int off = 16; off; off >>= 1) s += __shfl_xor(s, off, 32);
  if (lane == 0) out[g] = s / fmaxf((float)cnt[g], 1.f) + blin[0];
}

// ---------------- host side ----------------
static inline void* carve(char*& p, size_t bytes) {
  void* r = (void*)p;
  p += (bytes + 255) & ~(size_t)255;
  return r;
}

extern "C" void kernel_launch(void* const* d_in, const int* in_sizes, int n_in,
                              void* d_out, int out_size, void* d_ws, size_t ws_size,
                              hipStream_t stream) {
  const int N = in_sizes[0] / 43;       // 20000
  const int E = in_sizes[22] / 2;       // 320000
  const int G = out_size;               // 500

  const float* x    = (const float*)d_in[0];
  const float* ea   = (const float*)d_in[1];
  const float* W[3]     = { (const float*)d_in[2],  (const float*)d_in[8],  (const float*)d_in[14] };
  const float* We[3]    = { (const float*)d_in[3],  (const float*)d_in[9],  (const float*)d_in[15] };
  const float* asrc[3]  = { (const float*)d_in[4],  (const float*)d_in[10], (const float*)d_in[16] };
  const float* adst[3]  = { (const float*)d_in[5],  (const float*)d_in[11], (const float*)d_in[17] };
  const float* aedge[3] = { (const float*)d_in[6],  (const float*)d_in[12], (const float*)d_in[18] };
  const float* bias[3]  = { (const float*)d_in[7],  (const float*)d_in[13], (const float*)d_in[19] };
  const float* Wlin = (const float*)d_in[20];
  const float* blin = (const float*)d_in[21];
  const int* src  = (const int*)d_in[22];
  const int* dst  = src + E;
  const int* batch = (const int*)d_in[23];
  float* out = (float*)d_out;

  char* p = (char*)d_ws;
  float*     bufA      = (float*)    carve(p, (size_t)N * HC * 4);  // layer outputs
  float*     bufH      = (float*)    carve(p, (size_t)N * HC * 4);  // h = X@W scratch
  _Float16*  Ah        = (_Float16*) carve(p, (size_t)N * HC * 2);  // f16 activations
  _Float16*  Wh        = (_Float16*) carve(p, (size_t)HC * HC * 2); // f16 padded weights
  float*     aebuf     = (float*)    carve(p, (size_t)E * NH * 4);  // ae -> alpha -> ex
  float*     al        = (float*)    carve(p, (size_t)N * NH * 4);
  float*     ar        = (float*)    carve(p, (size_t)N * NH * 4);
  float*     loopsum   = (float*)    carve(p, (size_t)N * NH * 4);
  float*     loopalpha = (float*)    carve(p, (size_t)N * NH * 4);  // -> att_loop
  float*     denom     = (float*)    carve(p, (size_t)N * NH * 4);  // -> inv_denom
  unsigned*  amax      = (unsigned*) carve(p, (size_t)N * NH * 4);
  int*       deg       = (int*)      carve(p, (size_t)N * 4);
  int*       rowstart  = (int*)      carve(p, (size_t)(N + 1) * 4);
  int*       fillpos   = (int*)      carve(p, (size_t)N * 4);
  int*       csr       = (int*)      carve(p, (size_t)E * 4);
  float*     wfold     = (float*)    carve(p, 64 * 4);
  float*     pooled    = (float*)    carve(p, (size_t)G * HC * 4);
  int*       cnt       = (int*)      carve(p, (size_t)G * 4);
  (void)ws_size; (void)n_in;

  const int TB = 256;
  auto cdiv = [](int a, int b) { return (a + b - 1) / b; };

  // ---- CSR build ----
  hipMemsetAsync(deg, 0, (size_t)N * 4, stream);
  hipMemsetAsync(fillpos, 0, (size_t)N * 4, stream);
  count_deg<<<cdiv(E, TB), TB, 0, stream>>>(dst, deg, E);
  scan_deg<<<1, 1024, 0, stream>>>(deg, rowstart, N);
  csr_fill<<<cdiv(E, TB), TB, 0, stream>>>(dst, rowstart, fillpos, csr, E);

  const float* in_feats = x;
  for (int l = 0; l < 3; ++l) {
    int K  = (l == 0) ? 43 : HC;
    int Kp = (l == 0) ? 64 : HC;       // padded to multiple of 32
    int relu = (l < 2) ? 1 : 0;

    // f16 conversion (padded, guard-free WMMA loop)
    cvt_pad_a<<<cdiv(N * Kp, TB), TB, 0, stream>>>(in_feats, Ah, N, K, Kp);
    cvt_pad_w<<<cdiv(Kp * HC, TB), TB, 0, stream>>>(W[l], Wh, K, Kp, HC);

    // h = in_feats @ W : one wave per 16x64 strip
    int waves = (N / 16) * (HC / 64);
    gemm_wmma<<<cdiv(waves * 32, TB), TB, 0, stream>>>(Ah, Wh, bufH, N, Kp, HC);

    fold_we<<<1, 64, 0, stream>>>(We[l], aedge[l], wfold);
    node_alr<<<cdiv(N, TB / 32), TB, 0, stream>>>(bufH, asrc[l], adst[l], al, ar, N);

    hipMemsetAsync(loopsum, 0, (size_t)N * NH * 4, stream);
    edge_ae<<<cdiv(E, TB), TB, 0, stream>>>(ea, wfold, dst, aebuf, loopsum, E);
    node_loop<<<cdiv(N * NH, TB), TB, 0, stream>>>(al, ar, loopsum, deg, loopalpha, amax, denom, N * NH);
    edge_alpha<<<cdiv(E * NH, TB), TB, 0, stream>>>(al, ar, src, dst, aebuf, amax, E * NH);
    edge_exp<<<cdiv(E * NH, TB), TB, 0, stream>>>(aebuf, dst, amax, denom, E * NH);
    node_norm<<<cdiv(N * NH, TB), TB, 0, stream>>>(loopalpha, amax, denom, N * NH);

    aggregate<<<N, 160, 0, stream>>>(bufH, src, rowstart, csr, aebuf, denom,
                                     loopalpha, bias[l], bufA, relu, N);
    in_feats = bufA;
  }

  // ---- global mean pool + linear ----
  hipMemsetAsync(pooled, 0, (size_t)G * HC * 4, stream);
  hipMemsetAsync(cnt, 0, (size_t)G * 4, stream);
  pool_cnt<<<cdiv(N, TB), TB, 0, stream>>>(batch, cnt, N);
  pool_sum<<<cdiv(N * HC, TB), TB, 0, stream>>>(bufA, batch, pooled, N * HC);
  pool_final<<<cdiv(G, TB / 32), TB, 0, stream>>>(pooled, cnt, Wlin, blin, out, G);
}
